// SAMVisionAttention_10866267259108
// MI455X (gfx1250) — matrix operand order repaired and checked by reading, not compiled, not run on hardware
//
#include <hip/hip_runtime.h>

// ---------------------------------------------------------------------------
// Types and helpers
// ---------------------------------------------------------------------------
typedef unsigned short u16;
typedef __bf16 bf16_t;
typedef bf16_t v16bf __attribute__((ext_vector_type(16)));
typedef float  v8f   __attribute__((ext_vector_type(8)));

__device__ __forceinline__ u16 f2bf(float f) {
    union { float f; unsigned u; } a; a.f = f;
    unsigned u = a.u;
    u += 0x7FFFu + ((u >> 16) & 1u);          // round to nearest even
    return (u16)(u >> 16);
}
__device__ __forceinline__ float bf2f(u16 h) {
    union { unsigned u; float f; } a; a.u = ((unsigned)h) << 16;
    return a.f;
}
__device__ __forceinline__ v8f vzero8() {
    v8f z;
#pragma unroll
    for (int i = 0; i < 8; ++i) z[i] = 0.0f;
    return z;
}

union Frag {                     // one 16-half WMMA operand fragment per lane
    v16bf v;
    u16   s[16];
    uint4 q[2];
};

__device__ __forceinline__ v8f wmma_bf16(const v16bf& a, const v16bf& b, v8f c) {
    return __builtin_amdgcn_wmma_f32_16x16x32_bf16(
        /*neg_a=*/false, a, /*neg_b=*/false, b,
        /*c_mod=*/(short)0, c, /*reuse_a=*/false, /*reuse_b=*/false);
}

// CDNA5 async global->LDS copy (per-lane 16B), tracked by ASYNCcnt.
// LDS address operand: flat/generic LDS pointers carry the LDS byte offset in
// the low 32 bits (ISA 10.2).
__device__ __forceinline__ void async_b128(const void* gsrc, void* lds_dst) {
    asm volatile("global_load_async_to_lds_b128 %0, %1, off"
                 :: "v"((unsigned)(size_t)lds_dst), "v"(gsrc)
                 : "memory");
}
__device__ __forceinline__ void wait_async0() {
    asm volatile("s_wait_asynccnt 0x0" ::: "memory");
}

// Problem constants
#define NH   12
#define HD   64
#define SEQ  2304   // 48*48
#define CDIM 768

// ---------------------------------------------------------------------------
// Pack kernels (run once): bf16 conversion and transposes
// ---------------------------------------------------------------------------
__global__ __launch_bounds__(256)
void pack_bf16(const float* __restrict__ in, u16* __restrict__ out, int n)
{
    const int i = (blockIdx.x * 256 + threadIdx.x) * 4;
    if (i < n) {
#pragma unroll
        for (int j = 0; j < 4; ++j) out[i + j] = f2bf(in[i + j]);
    }
}

// Wt[n][k] = bf16(W[k][n]); 64x64 LDS tiles, both sides coalesced
__global__ __launch_bounds__(256)
void transpose_pack_w(const float* __restrict__ W, u16* __restrict__ Wt, int K, int N)
{
    __shared__ u16 sT[64 * 72];
    const int k0 = blockIdx.y * 64, n0 = blockIdx.x * 64;
    const int t = threadIdx.x;
#pragma unroll
    for (int i = 0; i < 16; ++i) {
        const int idx = t + i * 256;
        const int kk = idx >> 6, nn = idx & 63;
        sT[kk * 72 + nn] = f2bf(W[(size_t)(k0 + kk) * N + n0 + nn]);
    }
    __syncthreads();
#pragma unroll
    for (int i = 0; i < 16; ++i) {
        const int idx = t + i * 256;
        const int nn = idx >> 6, kk = idx & 63;
        Wt[(size_t)(n0 + nn) * K + k0 + kk] = sT[kk * 72 + nn];
    }
}

// Vt[head][d][s] = V[head][s][d]; per-head 64(s) x 64(d) tiles
__global__ __launch_bounds__(256)
void transpose_v(const u16* __restrict__ V, u16* __restrict__ Vt)
{
    __shared__ u16 sT[64 * 72];
    const int head = blockIdx.y;
    const int s0 = blockIdx.x * 64;
    const int t = threadIdx.x;
#pragma unroll
    for (int i = 0; i < 16; ++i) {
        const int idx = t + i * 256;
        const int ss = idx >> 6, dd = idx & 63;
        sT[ss * 72 + dd] = V[((size_t)head * SEQ + s0 + ss) * HD + dd];
    }
    __syncthreads();
#pragma unroll
    for (int i = 0; i < 16; ++i) {
        const int idx = t + i * 256;
        const int dd = idx >> 6, ss = idx & 63;
        Vt[((size_t)head * HD + dd) * SEQ + s0 + ss] = sT[ss * 72 + dd];
    }
}

// ---------------------------------------------------------------------------
// bf16 WMMA GEMM, 128x128 block tile, 8 waves of (4M x 2N) 16x16 frags.
// A: bf16 row-major (M x Kd).  B: bf16 pre-transposed Bt[n][k] (N x Kd).
// MODE 0: epilogue routes columns into per-head Q/K/V bf16 buffers (+bias)
// MODE 1: epilogue writes f32 out (+bias)
// ---------------------------------------------------------------------------
template <int MODE>
__global__ __launch_bounds__(256)
void gemm_wmma(const u16* __restrict__ Abf, const u16* __restrict__ Bt,
               const float* __restrict__ bias,
               u16* __restrict__ Qb, u16* __restrict__ Kb, u16* __restrict__ Vb,
               float* __restrict__ Of32, int Kd, int Nout)
{
    __shared__ __align__(16) u16 sA[128 * 40];   // A tile row-major, pitch 40 halfs
    __shared__ __align__(16) u16 sB[128 * 40];   // Bt tile row-major, pitch 40 halfs

    const int t   = threadIdx.x;
    const int wv  = t >> 5;
    const int ln  = t & 31;
    const int g   = ln >> 4;
    const int l16 = ln & 15;
    const int wm  = wv >> 2;        // wave row block (0..1)  -> 64 rows
    const int wn  = wv & 3;         // wave col block (0..3)  -> 32 cols
    const int m0  = blockIdx.y * 128;
    const int n0  = blockIdx.x * 128;

    v8f acc[4][2];
#pragma unroll
    for (int i = 0; i < 4; ++i)
#pragma unroll
        for (int j = 0; j < 2; ++j) acc[i][j] = vzero8();

    const int row = t >> 1;
    const int cs  = (t & 1) * 16;

    for (int k0 = 0; k0 < Kd; k0 += 32) {
        __syncthreads();
        {   // ---- async-stage A (128x32) and Bt (128x32), pure 16B copies ----
            const u16* asrc = Abf + (size_t)(m0 + row) * Kd + k0 + cs;
            async_b128(asrc,     &sA[row * 40 + cs]);
            async_b128(asrc + 8, &sA[row * 40 + cs + 8]);
            const u16* bsrc = Bt + (size_t)(n0 + row) * Kd + k0 + cs;
            async_b128(bsrc,     &sB[row * 40 + cs]);
            async_b128(bsrc + 8, &sB[row * 40 + cs + 8]);
        }
        wait_async0();
        __syncthreads();

        Frag a[4], b[2];
#pragma unroll
        for (int mf = 0; mf < 4; ++mf) {
            const int base = (wm * 64 + mf * 16 + l16) * 40;
            a[mf].q[0] = *(const uint4*)&sA[base + g * 8];        // K 0-7 / 8-15
            a[mf].q[1] = *(const uint4*)&sA[base + 16 + g * 8];   // K 16-23 / 24-31
        }
#pragma unroll
        for (int nf = 0; nf < 2; ++nf) {
            const int base = (wn * 32 + nf * 16 + l16) * 40;
            b[nf].q[0] = *(const uint4*)&sB[base + g * 8];
            b[nf].q[1] = *(const uint4*)&sB[base + 16 + g * 8];
        }
#pragma unroll
        for (int mf = 0; mf < 4; ++mf)
#pragma unroll
            for (int nf = 0; nf < 2; ++nf)
                acc[mf][nf] = wmma_bf16(a[mf].v, b[nf].v, acc[mf][nf]);
    }

    // ---- epilogue ----
#pragma unroll
    for (int nf = 0; nf < 2; ++nf) {
        const int col = n0 + wn * 32 + nf * 16 + l16;
        const float bv = bias[col];
#pragma unroll
        for (int mf = 0; mf < 4; ++mf) {
#pragma unroll
            for (int r = 0; r < 8; ++r) {
                const int rowm = m0 + wm * 64 + mf * 16 + r + 8 * g;
                const float val = acc[mf][nf][r] + bv;
                if constexpr (MODE == 0) {
                    const int which = col / CDIM;
                    const int rr    = col % CDIM;
                    const int head  = rr >> 6;
                    const int d     = rr & 63;
                    u16* dst = (which == 0) ? Qb : ((which == 1) ? Kb : Vb);
                    dst[((size_t)head * SEQ + rowm) * HD + d] = f2bf(val);
                } else {
                    Of32[(size_t)rowm * Nout + col] = val;
                }
            }
        }
    }
}

// ---------------------------------------------------------------------------
// Relative-position bias tables
//   RH[head][s][kh] = sum_d q[head][s][d] * rel_pos_h[h(s)-kh+47][d]
//   RW[head][s][kw] = sum_d q[head][s][d] * rel_pos_w[w(s)-kw+47][d]
// ---------------------------------------------------------------------------
__global__ __launch_bounds__(128)
void rel_bias_kernel(const u16* __restrict__ Qb,
                     const float* __restrict__ rel_h, const float* __restrict__ rel_w,
                     float* __restrict__ RH, float* __restrict__ RW)
{
    const int idx = blockIdx.x * blockDim.x + threadIdx.x;
    const int total = NH * SEQ * 96;
    if (idx >= total) return;
    const int j48  = idx % 96;
    const int tmp  = idx / 96;
    const int s    = tmp % SEQ;
    const int head = tmp / SEQ;
    const int isW  = (j48 >= 48);
    const int j    = j48 & 47;
    const int hq   = s / 48;
    const int wq   = s % 48;
    const int rrow = (isW ? (wq - j) : (hq - j)) + 47;

    const float* rp = (isW ? rel_w : rel_h) + (size_t)rrow * HD;
    const u16*   qp = Qb + ((size_t)head * SEQ + s) * HD;
    float acc = 0.0f;
#pragma unroll 8
    for (int d = 0; d < HD; ++d) acc += bf2f(qp[d]) * rp[d];
    float* dst = isW ? RW : RH;
    dst[((size_t)head * SEQ + s) * 48 + j] = acc;
}

// ---------------------------------------------------------------------------
// Fused flash attention with decomposed rel-pos bias
//   grid = (SEQ/64, NH), block = 128 (4 waves x 16 q-rows each)
// ---------------------------------------------------------------------------
__global__ __launch_bounds__(128)
void flash_attn_kernel(const u16* __restrict__ Qb, const u16* __restrict__ Kb,
                       const u16* __restrict__ Vt,
                       const float* __restrict__ RH, const float* __restrict__ RW,
                       u16* __restrict__ Ob)
{
    __shared__ __align__(16) u16 sK[32 * 72];     // K chunk row-major [key][d], pitch 72
    __shared__ __align__(16) u16 sVt[64 * 40];    // V chunk transposed [d][key], pitch 40
    __shared__ __align__(16) u16 sP[4 * 16 * 32]; // per-wave P tile (q x key) bf16
    __shared__ float sRH[64 * 48];
    __shared__ float sRW[64 * 48];

    const int t    = threadIdx.x;
    const int wv   = t >> 5;
    const int ln   = t & 31;
    const int g    = ln >> 4;
    const int l16  = ln & 15;
    const int head = blockIdx.y;
    const int q0   = blockIdx.x * 64;
    const int qw0  = q0 + wv * 16;

    // preload bias tables for this q-block (contiguous rows)
    {
        const float* rhs = RH + ((size_t)head * SEQ + q0) * 48;
        const float* rws = RW + ((size_t)head * SEQ + q0) * 48;
        for (int i = t; i < 64 * 48; i += 128) { sRH[i] = rhs[i]; sRW[i] = rws[i]; }
    }

    // Q fragments (A-matrix layout), kept for the whole kernel
    Frag qa[2];
    {
        const u16* qrow = Qb + ((size_t)head * SEQ + qw0 + l16) * HD;
#pragma unroll
        for (int f = 0; f < 2; ++f) {
            const int d0 = f * 32;
            qa[f].q[0] = *(const uint4*)(qrow + d0 + g * 8);
            qa[f].q[1] = *(const uint4*)(qrow + d0 + 16 + g * 8);
        }
    }

    v8f oacc[4];
#pragma unroll
    for (int f = 0; f < 4; ++f) oacc[f] = vzero8();
    float m8[8], l8[8];
#pragma unroll
    for (int r = 0; r < 8; ++r) { m8[r] = -3.0e38f; l8[r] = 0.0f; }

    const float scale = 0.125f;          // hd^-0.5, hd = 64
    const float LOG2E = 1.44269504f;

    // per-thread staging indices (constant across chunks)
    const int kidx = t * 16;
    const int krow = kidx >> 6;
    const int kd   = kidx & 63;
    const int vdd  = t >> 1;             // sVt row (d)
    const int vhf  = (t & 1) * 16;       // key half within row

    for (int kc = 0; kc < SEQ / 32; ++kc) {
        const int key0 = kc * 32;
        __syncthreads();
        {   // ---- async-stage K (row-major) and V (pre-transposed rows) ----
            const u16* ksrc = Kb + ((size_t)head * SEQ + key0) * HD + kidx;
            async_b128(ksrc,     &sK[krow * 72 + kd]);
            async_b128(ksrc + 8, &sK[krow * 72 + kd + 8]);
            const u16* vsrc = Vt + ((size_t)head * HD + vdd) * SEQ + key0 + vhf;
            async_b128(vsrc,     &sVt[vdd * 40 + vhf]);
            async_b128(vsrc + 8, &sVt[vdd * 40 + vhf + 8]);
        }
        wait_async0();
        __syncthreads();

        // ---- S = Q K^T (16 q x 32 keys, two 16x16 D-frags) ----
        v8f sfr[2];
#pragma unroll
        for (int c = 0; c < 2; ++c) {
            const int kr = c * 16 + l16;
            Frag kb[2];
#pragma unroll
            for (int f = 0; f < 2; ++f) {
                const int d0 = f * 32;
                kb[f].q[0] = *(const uint4*)&sK[kr * 72 + d0 + g * 8];
                kb[f].q[1] = *(const uint4*)&sK[kr * 72 + d0 + 16 + g * 8];
            }
            v8f z = vzero8();
            sfr[c] = wmma_bf16(qa[0].v, kb[0].v, z);
            sfr[c] = wmma_bf16(qa[1].v, kb[1].v, sfr[c]);
        }

        // ---- scale + rel-pos bias, online softmax ----
        float sv[2][8];
#pragma unroll
        for (int c = 0; c < 2; ++c) {
            const int kg = key0 + c * 16 + l16;
            const int kh = kg / 48;
            const int kw = kg % 48;
#pragma unroll
            for (int r = 0; r < 8; ++r) {
                const int rloc = wv * 16 + r + 8 * g;   // row within 64-row block
                sv[c][r] = sfr[c][r] * scale + sRH[rloc * 48 + kh] + sRW[rloc * 48 + kw];
            }
        }
        float mnew[8], alpha[8], p[2][8];
#pragma unroll
        for (int r = 0; r < 8; ++r) {
            float mx = fmaxf(sv[0][r], sv[1][r]);
#pragma unroll
            for (int msk = 1; msk < 16; msk <<= 1) mx = fmaxf(mx, __shfl_xor(mx, msk, 16));
            mnew[r]  = fmaxf(m8[r], mx);
            alpha[r] = exp2f((m8[r] - mnew[r]) * LOG2E);
            m8[r]    = mnew[r];
        }
#pragma unroll
        for (int c = 0; c < 2; ++c)
#pragma unroll
            for (int r = 0; r < 8; ++r) p[c][r] = exp2f((sv[c][r] - mnew[r]) * LOG2E);
#pragma unroll
        for (int r = 0; r < 8; ++r) {
            float rs = p[0][r] + p[1][r];
#pragma unroll
            for (int msk = 1; msk < 16; msk <<= 1) rs += __shfl_xor(rs, msk, 16);
            l8[r] = l8[r] * alpha[r] + rs;
        }
#pragma unroll
        for (int f = 0; f < 4; ++f)
#pragma unroll
            for (int r = 0; r < 8; ++r) oacc[f][r] *= alpha[r];

        // ---- route P (D-layout) -> A-layout via per-wave LDS blob ----
#pragma unroll
        for (int c = 0; c < 2; ++c)
#pragma unroll
            for (int r = 0; r < 8; ++r)
                sP[wv * 512 + (r + 8 * g) * 32 + c * 16 + l16] = f2bf(p[c][r]);
        __syncthreads();
        Frag pa;
        {
            const u16* pb = &sP[wv * 512 + l16 * 32];
            pa.q[0] = *(const uint4*)(pb + g * 8);
            pa.q[1] = *(const uint4*)(pb + 16 + g * 8);
        }
        // ---- O += P * V ----
#pragma unroll
        for (int f = 0; f < 4; ++f) {
            const int vrow = f * 16 + l16;
            Frag vb;
            vb.q[0] = *(const uint4*)&sVt[vrow * 40 + g * 8];
            vb.q[1] = *(const uint4*)&sVt[vrow * 40 + 16 + g * 8];
            oacc[f] = wmma_bf16(pa.v, vb.v, oacc[f]);
        }
    }

    // ---- finalize: divide by row sums, store O as bf16 (SEQ x 768) ----
#pragma unroll
    for (int r = 0; r < 8; ++r) {
        const float inv = 1.0f / l8[r];
        const int row = qw0 + r + 8 * g;
#pragma unroll
        for (int f = 0; f < 4; ++f) {
            const int col = head * HD + f * 16 + l16;
            Ob[(size_t)row * CDIM + col] = f2bf(oacc[f][r] * inv);
        }
    }
}

// ---------------------------------------------------------------------------
// Launch
// ---------------------------------------------------------------------------
extern "C" void kernel_launch(void* const* d_in, const int* in_sizes, int n_in,
                              void* d_out, int out_size, void* d_ws, size_t ws_size,
                              hipStream_t stream)
{
    (void)in_sizes; (void)n_in; (void)out_size; (void)ws_size;
    const float* x       = (const float*)d_in[0];
    const float* qkv_w   = (const float*)d_in[1];
    const float* qkv_b   = (const float*)d_in[2];
    const float* proj_w  = (const float*)d_in[3];
    const float* proj_b  = (const float*)d_in[4];
    const float* rel_h   = (const float*)d_in[5];
    const float* rel_w   = (const float*)d_in[6];

    char* ws = (char*)d_ws;
    const size_t QKV_BYTES = (size_t)NH * SEQ * HD * sizeof(u16);   // 3,538,944
    const size_t REL_BYTES = (size_t)NH * SEQ * 48 * sizeof(float); // 5,308,416
    size_t off = 0;
    u16*   Qb     = (u16*)(ws + off); off += QKV_BYTES;
    u16*   Kb     = (u16*)(ws + off); off += QKV_BYTES;
    u16*   Vb     = (u16*)(ws + off); off += QKV_BYTES;
    float* RH     = (float*)(ws + off); off += REL_BYTES;
    float* RW     = (float*)(ws + off); off += REL_BYTES;
    u16*   Ob     = (u16*)(ws + off); off += QKV_BYTES;             // SEQ x 768 bf16
    u16*   Xbf    = (u16*)(ws + off); off += (size_t)SEQ * CDIM * 2;
    u16*   Wqkvt  = (u16*)(ws + off); off += (size_t)(3 * CDIM) * CDIM * 2;
    u16*   Wprjt  = (u16*)(ws + off); off += (size_t)CDIM * CDIM * 2;
    u16*   Vt     = (u16*)(ws + off); off += QKV_BYTES;

    // 0) one-time packs: x -> bf16; weights -> bf16 transposed [n][k]
    pack_bf16<<<(SEQ * CDIM / 4 + 255) / 256, 256, 0, stream>>>(x, Xbf, SEQ * CDIM);
    transpose_pack_w<<<dim3(3 * CDIM / 64, CDIM / 64), 256, 0, stream>>>(
        qkv_w, Wqkvt, CDIM, 3 * CDIM);
    transpose_pack_w<<<dim3(CDIM / 64, CDIM / 64), 256, 0, stream>>>(
        proj_w, Wprjt, CDIM, CDIM);

    // 1) QKV projection -> per-head Q/K/V bf16
    gemm_wmma<0><<<dim3(2304 / 128, SEQ / 128), 256, 0, stream>>>(
        Xbf, Wqkvt, qkv_b, Qb, Kb, Vb, nullptr, CDIM, 2304);

    // 1b) V -> Vt[head][d][s] for contiguous flash staging
    transpose_v<<<dim3(SEQ / 64, NH), 256, 0, stream>>>(Vb, Vt);

    // 2) rel-pos bias tables RH/RW
    {
        const int total = NH * SEQ * 96;
        rel_bias_kernel<<<(total + 127) / 128, 128, 0, stream>>>(Qb, rel_h, rel_w, RH, RW);
    }

    // 3) fused attention
    flash_attn_kernel<<<dim3(SEQ / 64, NH), 128, 0, stream>>>(Qb, Kb, Vt, RH, RW, Ob);

    // 4) output projection -> f32 out
    gemm_wmma<1><<<dim3(CDIM / 128, SEQ / 128), 256, 0, stream>>>(
        Ob, Wprjt, proj_b, nullptr, nullptr, nullptr, (float*)d_out, CDIM, CDIM);
}
